// Attention_81604378624294
// MI455X (gfx1250) — matrix operand order repaired and checked by reading, not compiled, not run on hardware
//
#include <hip/hip_runtime.h>

// ---------------------------------------------------------------------------
// MI455X (gfx1250) fused attention block, bf16 WMMA + async LDS staging.
//   x[16,1024,768] @ w_qkv[2304,768]^T -> q,k,v [16,12,1024,64]
//   flash-style attention per (b,h), scale = 1/8
//   attn_out[16,1024,768] @ w_proj[768,768]^T + b_proj -> out (fp32)
// Data movement: GLOBAL_LOAD_ASYNC_TO_LDS_B128 (ASYNCcnt) double-buffered
// against v_wmma_f32_16x16x32_bf16 compute; global_prefetch for L2 warmup.
// ---------------------------------------------------------------------------

typedef __bf16 bf16;
typedef __attribute__((ext_vector_type(16))) __bf16 v16bf;
typedef __attribute__((ext_vector_type(8)))  __bf16 v8bf;
typedef __attribute__((ext_vector_type(8)))  float  v8f;

static_assert(sizeof(bf16) == 2, "bf16 must be 2 bytes");

#define WMMA_BF16(A, B, C) \
  __builtin_amdgcn_wmma_f32_16x16x32_bf16(false, (A), false, (B), (short)0, (C), false, false)

// async copy 16B: LDS[ldsOff] <- base[byteOff], tracked by ASYNCcnt
#define ASYNC_LDS_B128(ldsOff, byteOff, base)                                  \
  asm volatile("global_load_async_to_lds_b128 %0, %1, %2"                      \
               :: "v"(ldsOff), "v"(byteOff), "s"(base) : "memory")
#define WAIT_ASYNC0() asm volatile("s_wait_asynccnt 0x0" ::: "memory")

__device__ __forceinline__ unsigned lds_off(const void* p) {
  // generic LDS address: low 32 bits are the raw LDS byte offset (ISA 10.2)
  return (unsigned)(unsigned long long)p;
}

__device__ __forceinline__ v16bf cat16(v8bf lo, v8bf hi) {
  return __builtin_shufflevector(lo, hi, 0,1,2,3,4,5,6,7,8,9,10,11,12,13,14,15);
}

__device__ __forceinline__ float red_max16(float v) {
#pragma unroll
  for (int m = 1; m < 16; m <<= 1) v = fmaxf(v, __shfl_xor(v, m, 32));
  return v;  // xor masks 1..8 stay inside each 16-lane half of the wave32
}
__device__ __forceinline__ float red_sum16(float v) {
#pragma unroll
  for (int m = 1; m < 16; m <<= 1) v += __shfl_xor(v, m, 32);
  return v;
}

// ---------------------------------------------------------------------------
// fp32 -> bf16 conversion (grid-stride)
// ---------------------------------------------------------------------------
__global__ void cvt_f32_bf16(const float* __restrict__ s, bf16* __restrict__ d, int n) {
  int i = blockIdx.x * blockDim.x + threadIdx.x;
  int stride = gridDim.x * blockDim.x;
  for (; i < n; i += stride) d[i] = (bf16)s[i];
}

// ---------------------------------------------------------------------------
// Tiled bf16 GEMM:  out[m,e] = sum_k A[m,k] * W[e,k]   (M=16384, K=768)
// Block tile 128x128, 8 waves (2 M-halves x 4 N-quarters), wave = 64x32,
// 4x2 grid of v_wmma_f32_16x16x32_bf16 per 32-deep K step.
// Double-buffered LDS filled by global_load_async_to_lds_b128.
// mode 0: scatter bf16 into qkv layout [3][16][12][1024][64]
// mode 1: fp32 + bias into d_out [16384,768]
// ---------------------------------------------------------------------------
__global__ __launch_bounds__(256) void gemm_bf16(
    const bf16* __restrict__ A,       // [16384, 768] row-major
    const bf16* __restrict__ W,       // [E, 768] row-major
    bf16* __restrict__ qkv_dst,       // mode 0
    float* __restrict__ f_dst,        // mode 1
    const float* __restrict__ bias,   // mode 1
    int mode) {
  __shared__ bf16 As[2][128 * 32];    // [m][k] row-major, 8 KB per stage
  __shared__ bf16 Ws[2][128 * 32];    // [e][k] row-major == B^T, 8 KB per stage

  const int tid  = threadIdx.x;
  const int lane = tid & 31;
  const int wv   = tid >> 5;
  const int ln   = lane & 15;         // row/col within 16x16 tile
  const int hf   = lane >> 4;         // which 16-lane half
  const int wm   = wv & 1;            // M half (64 rows)
  const int wn   = wv >> 1;           // N quarter (32 cols)
  const int rowBase = blockIdx.y * 128;
  const int colBase = blockIdx.x * 128;

  // issue the async fill of one 128x32 A tile + 128x32 W tile into stage `st`
  auto issueTiles = [&](int kt, int st) {
#pragma unroll
    for (int i = 0; i < 2; ++i) {
      int c = tid + 256 * i;
      int r = c >> 2;
      int ko = (c & 3) * 8;
      unsigned lA = lds_off(&As[st][r * 32 + ko]);
      unsigned oA = (unsigned)(((rowBase + r) * 768 + kt + ko) * 2);
      ASYNC_LDS_B128(lA, oA, A);
      unsigned lW = lds_off(&Ws[st][r * 32 + ko]);
      unsigned oW = (unsigned)(((colBase + r) * 768 + kt + ko) * 2);
      ASYNC_LDS_B128(lW, oW, W);
    }
  };

  const v8f vzero = {0.f, 0.f, 0.f, 0.f, 0.f, 0.f, 0.f, 0.f};
  v8f acc[4][2];
#pragma unroll
  for (int mt = 0; mt < 4; ++mt)
#pragma unroll
    for (int nt = 0; nt < 2; ++nt) acc[mt][nt] = vzero;

  issueTiles(0, 0);
  WAIT_ASYNC0();
  __syncthreads();

  int buf = 0;
  for (int kt = 0; kt < 768; kt += 32) {
    if (kt + 32 < 768) issueTiles(kt + 32, buf ^ 1);   // overlap with WMMAs
    if (kt + 64 < 768) {                               // L2 warmup 2 steps out
      __builtin_prefetch(&A[(size_t)(rowBase + (tid >> 1)) * 768 + kt + 64], 0, 1);
      __builtin_prefetch(&W[(size_t)(colBase + (tid >> 1)) * 768 + kt + 64], 0, 1);
    }

    // A fragments: lane ln holds row m; 16 contiguous K at offset 16h
    v16bf af[4];
#pragma unroll
    for (int mt = 0; mt < 4; ++mt) {
      int r = 64 * wm + 16 * mt + ln;
      v8bf lo = *(const v8bf*)&As[buf][r * 32 + 16 * hf];
      v8bf hi = *(const v8bf*)&As[buf][r * 32 + 16 * hf + 8];
      af[mt] = cat16(lo, hi);
    }
    // B fragments: lane ln = column n, 16 contiguous K at offset 16h
    v16bf bfr[2];
#pragma unroll
    for (int nt = 0; nt < 2; ++nt) {
      int n = 32 * wn + 16 * nt + ln;
      bfr[nt] = *(const v16bf*)&Ws[buf][n * 32 + 16 * hf];
    }
#pragma unroll
    for (int mt = 0; mt < 4; ++mt)
#pragma unroll
      for (int nt = 0; nt < 2; ++nt)
        acc[mt][nt] = WMMA_BF16(af[mt], bfr[nt], acc[mt][nt]);

    WAIT_ASYNC0();       // next stage fully in LDS
    __syncthreads();
    buf ^= 1;
  }

  // Epilogue. C layout: element r of lane -> row r + 8*hf, col ln (per tile).
#pragma unroll
  for (int mt = 0; mt < 4; ++mt)
#pragma unroll
    for (int nt = 0; nt < 2; ++nt)
#pragma unroll
      for (int r = 0; r < 8; ++r) {
        int row = rowBase + 64 * wm + 16 * mt + r + 8 * hf;
        int col = colBase + 32 * wn + 16 * nt + ln;
        float v = acc[mt][nt][r];
        if (mode == 0) {
          int s   = col / 768;           // 0:q 1:k 2:v
          int rem = col - s * 768;
          int hh  = rem >> 6;            // head
          int d   = rem & 63;
          int b   = row >> 10;
          int n   = row & 1023;
          qkv_dst[((((size_t)s * 16 + b) * 12 + hh) * 1024 + n) * 64 + d] = (bf16)v;
        } else {
          f_dst[(size_t)row * 768 + col] = v + bias[col];
        }
      }
}

// ---------------------------------------------------------------------------
// Fused flash attention. Grid = (16*12)*(1024/128) blocks, 256 threads.
// Each wave owns 16 query rows; streaming softmax over 16 key blocks of 64.
// K blocks staged by async-to-LDS; V transposed through VGPRs; both
// double-buffered against the WMMA pipeline.
// ---------------------------------------------------------------------------
__global__ __launch_bounds__(256) void attn_fused(
    const bf16* __restrict__ qkv,   // [3][16][12][1024][64]
    bf16* __restrict__ aout) {      // [16][1024][768]
  __shared__ bf16 Ks[2][64 * 64];   // K block, [key][d]   (B-layout for Q*K^T)
  __shared__ bf16 Vt[2][64 * 64];   // V block transposed, [d][key] (B for P*V)
  __shared__ bf16 Ps[8 * 16 * 64];  // per-wave P staging (C-layout -> A-layout)

  const int tid  = threadIdx.x;
  const int lane = tid & 31;
  const int wv   = tid >> 5;
  const int ln   = lane & 15;
  const int hf   = lane >> 4;
  const int qblk = blockIdx.x & 7;
  const int bh   = blockIdx.x >> 3;
  const int b    = bh / 12;
  const int h    = bh % 12;

  const size_t hs = (size_t)1024 * 64;
  const bf16* Qp = qkv + ((size_t)(0 * 16 + b) * 12 + h) * hs;
  const bf16* Kp = qkv + ((size_t)(1 * 16 + b) * 12 + h) * hs;
  const bf16* Vp = qkv + ((size_t)(2 * 16 + b) * 12 + h) * hs;

  auto issueK = [&](int kb, int st) {
#pragma unroll
    for (int i = 0; i < 2; ++i) {
      int c = tid + 256 * i;
      int r = c >> 3;
      int off = (c & 7) * 8;
      unsigned l = lds_off(&Ks[st][r * 64 + off]);
      unsigned g = (unsigned)(((kb * 64 + r) * 64 + off) * 2);
      ASYNC_LDS_B128(l, g, Kp);
    }
  };
  auto loadV = [&](int kb, v8bf* vr) {
#pragma unroll
    for (int i = 0; i < 2; ++i) {
      int c = tid + 256 * i;
      int r = c >> 3;
      int off = (c & 7) * 8;
      vr[i] = *(const v8bf*)&Vp[(size_t)(kb * 64 + r) * 64 + off];
    }
  };
  auto storeVt = [&](int st, const v8bf* vr) {
#pragma unroll
    for (int i = 0; i < 2; ++i) {
      int c = tid + 256 * i;
      int r = c >> 3;
      int off = (c & 7) * 8;
#pragma unroll
      for (int j = 0; j < 8; ++j) Vt[st][(off + j) * 64 + r] = vr[i][j];
    }
  };

  const int q0 = qblk * 128 + wv * 16;

  // Q fragments (fixed for whole kernel): two 32-deep K steps over d=64
  v16bf qf[2];
#pragma unroll
  for (int ks = 0; ks < 2; ++ks) {
    const bf16* qr = Qp + (size_t)(q0 + ln) * 64 + 32 * ks;
    v8bf lo = *(const v8bf*)(qr + 16 * hf);
    v8bf hi = *(const v8bf*)(qr + 16 * hf + 8);
    qf[ks] = cat16(lo, hi);
  }

  const v8f vzero = {0.f, 0.f, 0.f, 0.f, 0.f, 0.f, 0.f, 0.f};
  v8f o[4];
#pragma unroll
  for (int dt = 0; dt < 4; ++dt) o[dt] = vzero;
  float m_run[8], l_run[8];
#pragma unroll
  for (int r = 0; r < 8; ++r) { m_run[r] = -3.0e38f; l_run[r] = 0.f; }

  // prologue: stage block 0
  {
    issueK(0, 0);
    v8bf v0[2];
    loadV(0, v0);
    storeVt(0, v0);
    WAIT_ASYNC0();
    __syncthreads();
  }

  int buf = 0;
  for (int kb = 0; kb < 16; ++kb) {
    const bool more = (kb + 1) < 16;
    v8bf vnext[2];
    if (more) {               // overlap next block's traffic with this compute
      issueK(kb + 1, buf ^ 1);
      loadV(kb + 1, vnext);
    }

    // S = Q * K^T : 4 key tiles x 2 K-steps
    v8f s[4];
#pragma unroll
    for (int nt = 0; nt < 4; ++nt) s[nt] = vzero;
#pragma unroll
    for (int nt = 0; nt < 4; ++nt)
#pragma unroll
      for (int ks = 0; ks < 2; ++ks) {
        v16bf bfr = *(const v16bf*)&Ks[buf][(16 * nt + ln) * 64 + 32 * ks + 16 * hf];
        s[nt] = WMMA_BF16(qf[ks], bfr, s[nt]);
      }

    // scale
#pragma unroll
    for (int nt = 0; nt < 4; ++nt)
#pragma unroll
      for (int r = 0; r < 8; ++r) s[nt][r] *= 0.125f;

    // online softmax; C-layout: element r of lane -> query row r + 8*hf
#pragma unroll
    for (int r = 0; r < 8; ++r) {
      float mx = fmaxf(fmaxf(s[0][r], s[1][r]), fmaxf(s[2][r], s[3][r]));
      mx = red_max16(mx);
      float mnew  = fmaxf(m_run[r], mx);
      float alpha = __expf(m_run[r] - mnew);
      m_run[r] = mnew;
      float rs = 0.f;
#pragma unroll
      for (int nt = 0; nt < 4; ++nt) {
        float p = __expf(s[nt][r] - mnew);
        s[nt][r] = p;
        rs += p;
      }
      rs = red_sum16(rs);
      l_run[r] = l_run[r] * alpha + rs;
#pragma unroll
      for (int dt = 0; dt < 4; ++dt) o[dt][r] *= alpha;
      // stage P (bf16) into per-wave LDS region in [qrow][key] order
#pragma unroll
      for (int nt = 0; nt < 4; ++nt)
        Ps[wv * 1024 + (r + 8 * hf) * 64 + 16 * nt + ln] = (bf16)s[nt][r];
    }

    // O += P * V : A from Ps (per-wave, no barrier needed), B from Vt
#pragma unroll
    for (int ks = 0; ks < 2; ++ks) {
      const bf16* pr = &Ps[wv * 1024 + ln * 64 + 32 * ks];
      v8bf lo = *(const v8bf*)(pr + 16 * hf);
      v8bf hi = *(const v8bf*)(pr + 16 * hf + 8);
      v16bf af = cat16(lo, hi);
#pragma unroll
      for (int dt = 0; dt < 4; ++dt) {
        v16bf bfr = *(const v16bf*)&Vt[buf][(16 * dt + ln) * 64 + 32 * ks + 16 * hf];
        o[dt] = WMMA_BF16(af, bfr, o[dt]);
      }
    }

    if (more) storeVt(buf ^ 1, vnext);  // writes only the idle stage
    WAIT_ASYNC0();
    __syncthreads();
    buf ^= 1;
  }

  // normalize and write [b][n][h*64+d] as bf16 for the projection GEMM
#pragma unroll
  for (int r = 0; r < 8; ++r) {
    float inv = 1.f / l_run[r];
    int qrow = q0 + r + 8 * hf;
    size_t base = ((size_t)b * 1024 + qrow) * 768 + h * 64;
#pragma unroll
    for (int dt = 0; dt < 4; ++dt)
      aout[base + 16 * dt + ln] = (bf16)(o[dt][r] * inv);
  }
}

// ---------------------------------------------------------------------------
// Host-side launcher
// ---------------------------------------------------------------------------
extern "C" void kernel_launch(void* const* d_in, const int* in_sizes, int n_in,
                              void* d_out, int out_size, void* d_ws, size_t ws_size,
                              hipStream_t stream) {
  const float* x      = (const float*)d_in[0];
  const float* w_qkv  = (const float*)d_in[1];
  const float* w_proj = (const float*)d_in[2];
  const float* b_proj = (const float*)d_in[3];
  float* out = (float*)d_out;

  // workspace layout (bf16), ~131 MB total
  char* p = (char*)d_ws;
  bf16* x_bf   = (bf16*)p; p += (size_t)16384 * 768 * 2;             // 24 MB
  bf16* wq_bf  = (bf16*)p; p += (size_t)2304 * 768 * 2;              // 3.4 MB
  bf16* wp_bf  = (bf16*)p; p += (size_t)768 * 768 * 2;               // 1.1 MB
  bf16* qkv_bf = (bf16*)p; p += (size_t)3 * 16 * 12 * 1024 * 64 * 2; // 72 MB
  bf16* ao_bf  = (bf16*)p; p += (size_t)16384 * 768 * 2;             // 24 MB
  (void)ws_size; (void)in_sizes; (void)n_in; (void)out_size;

  cvt_f32_bf16<<<2048, 256, 0, stream>>>(x, x_bf, 16384 * 768);
  cvt_f32_bf16<<<1024, 256, 0, stream>>>(w_qkv, wq_bf, 2304 * 768);
  cvt_f32_bf16<<<512, 256, 0, stream>>>(w_proj, wp_bf, 768 * 768);

  // QKV projection: [16384,768] x [2304,768]^T -> scattered q/k/v
  gemm_bf16<<<dim3(2304 / 128, 16384 / 128), 256, 0, stream>>>(
      x_bf, wq_bf, qkv_bf, nullptr, nullptr, /*mode=*/0);

  // fused flash attention: 192 (b,h) pairs x 8 query slabs
  attn_fused<<<16 * 12 * 8, 256, 0, stream>>>(qkv_bf, ao_bf);

  // output projection with bias: fp32 out
  gemm_bf16<<<dim3(768 / 128, 16384 / 128), 256, 0, stream>>>(
      ao_bf, wp_bf, nullptr, out, b_proj, /*mode=*/1);
}
